// SGNet_CVAE_75746043232938
// MI455X (gfx1250) — matrix operand order repaired and checked by reading, not compiled, not run on hardware
//
#include <hip/hip_runtime.h>
#include <hip/hip_bf16.h>

typedef __attribute__((ext_vector_type(16))) _Float16 v16h;
typedef __attribute__((ext_vector_type(8)))  _Float16 v8h;
typedef __attribute__((ext_vector_type(8)))  float    v8f;

#define B_    32
#define ENC_  15
#define DEC_  45
#define H_    256
#define H4_   64
#define K_    20
#define L_    32
#define PRED_ 4
#define POSE_ 36

// d_out offsets (floats)
#define OUT_GOAL 0
#define OUT_CVAE 86400
#define OUT_KLD  1814400
#define OUT_PROB 1814401

// f16 weight-fragment region (offsets in halfs). Each 32x16 f16 B-fragment = 512 halfs.
#define F_DH2I  0
#define F_WIH   (F_DH2I + 8*16*512)
#define F_WHH   (F_WIH  + 8*48*512)
#define F_TOTAL (F_WHH  + 8*48*512)

// f32 workspace region (offsets in floats, region starts at byte F_TOTAL*2)
#define WS_TRAJIN 0
#define WS_ADDL   (WS_TRAJIN + B_*ENC_*H_)
#define WS_GA     (WS_ADDL   + B_*ENC_*H4_)
#define WS_GATT   (WS_GA     + ENC_*B_*768)
#define WS_DECH0  (WS_GATT   + ENC_*DEC_*B_*H4_)

__device__ __forceinline__ float sigmf(float x) { return 1.0f / (1.0f + __expf(-x)); }

__device__ __forceinline__ v8f wmma_f16(v16h a, v16h b, v8f c) {
  return __builtin_amdgcn_wmma_f32_16x16x32_f16(false, a, false, b, (short)0, c, false, false);
}

// A-fragment (16x32 f16, ISA 7.12.2): lane L -> row M=(L&15); halfs 0..7 cover
// K = ((L>>4)<<3)+0..7, halfs 8..15 cover K = 16+((L>>4)<<3)+0..7. Two 16B LDS reads.
__device__ __forceinline__ v16h load_a_frag(const _Float16* smem, int mBase, int kBase,
                                            int ld, int lane) {
  union { v16h v; v8h h[2]; } u;
  const _Float16* p = smem + (mBase + (lane & 15)) * ld + kBase + ((lane >> 4) << 3);
  u.h[0] = *(const v8h*)(p);
  u.h[1] = *(const v8h*)(p + 16);
  return u.v;
}

// ---------------- K0: swizzle decoder GEMM weights into f16 B-fragments ----------------
// B-fragment layout: lane L holds N=(L&15); element h holds K=((L>>4)<<3)+(h&7)+((h>>3)<<4).
__global__ void k0_fragw(const float* __restrict__ dh2iW, const float* __restrict__ decWih,
                         const float* __restrict__ decWhh, _Float16* __restrict__ fr) {
  for (int idx = blockIdx.x * blockDim.x + threadIdx.x; idx < F_TOTAL;
       idx += gridDim.x * blockDim.x) {
    const float* W; int ntiles, ldN, rowOff, base;
    if (idx < F_WIH)      { W = dh2iW;  ntiles = 16; ldN = 256; rowOff = 0;  base = idx; }
    else if (idx < F_WHH) { W = decWih; ntiles = 48; ldN = 768; rowOff = 64; base = idx - F_WIH; }
    else                  { W = decWhh; ntiles = 48; ldN = 768; rowOff = 0;  base = idx - F_WHH; }
    int f = base >> 9, q = base & 511;
    int lane = q >> 4, h = q & 15;
    int kt = f / ntiles, nt = f - kt * ntiles;
    int k = kt * 32 + ((lane >> 4) << 3) + (h & 7) + ((h >> 3) << 4);
    int n = nt * 16 + (lane & 15);
    fr[idx] = (_Float16)W[(size_t)(k + rowOff) * ldN + n];
  }
}

// ---------------- K1: per-(b,t) feature extraction, addl GRU(h=0), ga, prob/KLD --------
__global__ void __launch_bounds__(256)
k1_features(const float* __restrict__ inputs, const float* __restrict__ angle,
            const float* __restrict__ feW, const float* __restrict__ feB,
            const float* __restrict__ poseW, const float* __restrict__ poseB,
            const float* __restrict__ aWih, const float* __restrict__ aBih,
            const float* __restrict__ aBhh,
            const float* __restrict__ decWih, const float* __restrict__ decBih,
            float* __restrict__ trajin, float* __restrict__ addl, float* __restrict__ ga,
            float* __restrict__ out) {
  __shared__ float af[H4_], ad[H4_];
  int bt = blockIdx.x, b = bt / ENC_, t = bt - b * ENC_;
  int tid = threadIdx.x;
  const float* xi = inputs + (size_t)bt * PRED_;
  const float* an = angle + (size_t)bt * POSE_;
  if (tid < H4_) {
    float a = poseB[tid];
    for (int q = 0; q < POSE_; ++q) a += an[q] * poseW[q * H4_ + tid];
    af[tid] = a > 0.f ? a : 0.f;
  }
  {
    float a = feB[tid];
    for (int q = 0; q < PRED_; ++q) a += xi[q] * feW[q * H_ + tid];
    trajin[(size_t)bt * H_ + tid] = a > 0.f ? a : 0.f;
  }
  __syncthreads();
  if (tid < H4_) {
    float gr = aBih[tid], gz = aBih[H4_ + tid], gn = aBih[2 * H4_ + tid];
    for (int q = 0; q < H4_; ++q) {
      float x = af[q]; const float* w = aWih + q * 3 * H4_;
      gr += x * w[tid]; gz += x * w[H4_ + tid]; gn += x * w[2 * H4_ + tid];
    }
    float r = sigmf(gr + aBhh[tid]);
    float z = sigmf(gz + aBhh[H4_ + tid]);
    float n = tanhf(gn + r * aBhh[2 * H4_ + tid]);
    float o = (1.f - z) * n;                 // hidden = 0
    ad[tid] = o;
    addl[(size_t)bt * H4_ + tid] = o;
  }
  __syncthreads();
  for (int col = tid; col < 768; col += 256) {   // ga = addl @ Wih[320:384] + bih
    float a = decBih[col];
    for (int q = 0; q < H4_; ++q) a += ad[q] * decWih[(size_t)(320 + q) * 768 + col];
    ga[(size_t)(t * B_ + b) * 768 + col] = a;
  }
  if (tid < K_) out[OUT_PROB + (size_t)bt * K_ + tid] = 1.0f / K_;
  if (bt == 0 && tid == 0) out[OUT_KLD] = 0.0f;
}

// ---------------- K2: sequential encoder chain, one WG per batch element ---------------
struct EncP {
  const float *teWih, *teWhh, *teBih, *teBhh;
  const float *e2gW, *e2gB;
  const float *gWih, *gWhh, *gBih, *gBhh;
  const float *gh2iW, *gh2iB;
  const float *g2tW, *g2tB;
  const float *regW, *regB;
  const float *g2eW, *g2eB, *eaW, *eaB;
  const float *g2dW, *g2dB, *daW, *daB;
  const float *e2dW, *e2dB;
  const float *prW, *prB;
  const float *cvW, *cvB;
  const float *eps;
  const float *trajin;
  float *gatt, *dech0, *out;
};

__device__ __forceinline__ void softmax45(float* sc, float* red, int tid) {
  if (tid == 0) { float m = sc[0]; for (int s = 1; s < DEC_; ++s) m = fmaxf(m, sc[s]); red[0] = m; }
  __syncthreads();
  if (tid < DEC_) sc[tid] = __expf(sc[tid] - red[0]);
  __syncthreads();
  if (tid == 0) { float s0 = 0.f; for (int s = 0; s < DEC_; ++s) s0 += sc[s]; red[1] = s0; }
  __syncthreads();
  if (tid < DEC_) sc[tid] = sc[tid] / red[1];
  __syncthreads();
}

__global__ void __launch_bounds__(256) k2_encoder(EncP P) {
  __shared__ float xin[H_ + H4_];
  __shared__ float trajh[H_], newh[H_];
  __shared__ float gfe[H4_];
  __shared__ float ghid[H4_], gin[H4_], gnew[H4_];
  __shared__ float goals[DEC_ * H4_];
  __shared__ float Gw[DEC_ * H4_];
  __shared__ float gtv[H_];
  __shared__ float dh0[H_], hpart[H_];
  __shared__ float zk[K_ * L_];
  __shared__ float muv[L_], sigv[L_];
  __shared__ float sc[DEC_], red[2];
  int b = blockIdx.x, tid = threadIdx.x;
  trajh[tid] = 0.f;
  if (tid < H4_) gfe[tid] = 0.f;
  __syncthreads();
  for (int t = 0; t < ENC_; ++t) {
    xin[tid] = P.trajin[(size_t)(b * ENC_ + t) * H_ + tid];
    if (tid < H4_) xin[H_ + tid] = gfe[tid];
    __syncthreads();
    {  // traj_enc GRU
      int j = tid;
      float gr = P.teBih[j], gz = P.teBih[H_ + j], gn = P.teBih[2 * H_ + j];
      for (int q = 0; q < H_ + H4_; ++q) {
        float x = xin[q]; const float* w = P.teWih + (size_t)q * 3 * H_;
        gr += x * w[j]; gz += x * w[H_ + j]; gn += x * w[2 * H_ + j];
      }
      float hr = P.teBhh[j], hz = P.teBhh[H_ + j], hn = P.teBhh[2 * H_ + j];
      for (int q = 0; q < H_; ++q) {
        float x = trajh[q]; const float* w = P.teWhh + (size_t)q * 3 * H_;
        hr += x * w[j]; hz += x * w[H_ + j]; hn += x * w[2 * H_ + j];
      }
      float r = sigmf(gr + hr), z = sigmf(gz + hz);
      float n = tanhf(gn + r * hn);
      newh[j] = (1.f - z) * n + z * trajh[j];
    }
    __syncthreads();
    trajh[tid] = newh[tid];
    __syncthreads();
    if (tid < H4_) {  // goal hidden
      float a = P.e2gB[tid];
      for (int q = 0; q < H_; ++q) a += trajh[q] * P.e2gW[q * H4_ + tid];
      ghid[tid] = a > 0.f ? a : 0.f;
      gin[tid] = 0.f;
    }
    __syncthreads();
    for (int s = 0; s < DEC_; ++s) {  // SGE rollout
      if (tid < H4_) {
        int j = tid;
        float gr = P.gBih[j], gz = P.gBih[H4_ + j], gn = P.gBih[2 * H4_ + j];
        for (int q = 0; q < H4_; ++q) {
          float x = gin[q]; const float* w = P.gWih + q * 3 * H4_;
          gr += x * w[j]; gz += x * w[H4_ + j]; gn += x * w[2 * H4_ + j];
        }
        float hr = P.gBhh[j], hz = P.gBhh[H4_ + j], hn = P.gBhh[2 * H4_ + j];
        for (int q = 0; q < H4_; ++q) {
          float x = ghid[q]; const float* w = P.gWhh + q * 3 * H4_;
          hr += x * w[j]; hz += x * w[H4_ + j]; hn += x * w[2 * H4_ + j];
        }
        float r = sigmf(gr + hr), z = sigmf(gz + hz);
        float n = tanhf(gn + r * hn);
        gnew[j] = (1.f - z) * n + z * ghid[j];
      }
      __syncthreads();
      if (tid < H4_) { ghid[tid] = gnew[tid]; goals[s * H4_ + tid] = gnew[tid]; }
      __syncthreads();
      if (tid < H4_) {
        float a = P.gh2iB[tid];
        for (int q = 0; q < H4_; ++q) a += ghid[q] * P.gh2iW[q * H4_ + tid];
        gin[tid] = a > 0.f ? a : 0.f;
      }
      __syncthreads();
    }
    for (int s = 0; s < DEC_; ++s) {  // goal trajectory output
      {
        float a = P.g2tB[tid];
        for (int q = 0; q < H4_; ++q) a += goals[s * H4_ + q] * P.g2tW[q * H_ + tid];
        gtv[tid] = a > 0.f ? a : 0.f;
      }
      __syncthreads();
      if (tid < PRED_) {
        float a = P.regB[tid];
        for (int q = 0; q < H_; ++q) a += gtv[q] * P.regW[q * PRED_ + tid];
        P.out[OUT_GOAL + (size_t)((b * ENC_ + t) * DEC_ + s) * PRED_ + tid] = tanhf(a);
      }
      __syncthreads();
    }
    for (int w = tid; w < DEC_ * H4_; w += 256) {  // G_enc
      int s = w >> 6, j = w & 63;
      float a = P.g2eB[j];
      for (int q = 0; q < H4_; ++q) a += goals[s * H4_ + q] * P.g2eW[q * H4_ + j];
      Gw[w] = a > 0.f ? a : 0.f;
    }
    __syncthreads();
    if (tid < DEC_) {
      float a = P.eaB[0];
      for (int j = 0; j < H4_; ++j) a += tanhf(Gw[tid * H4_ + j]) * P.eaW[j];
      sc[tid] = a > 0.f ? a : 0.f;
    }
    __syncthreads();
    softmax45(sc, red, tid);
    if (tid < H4_) {
      float a = 0.f;
      for (int s = 0; s < DEC_; ++s) a += sc[s] * Gw[s * H4_ + tid];
      gfe[tid] = a;
    }
    __syncthreads();
    for (int w = tid; w < DEC_ * H4_; w += 256) {  // G_dec
      int s = w >> 6, j = w & 63;
      float a = P.g2dB[j];
      for (int q = 0; q < H4_; ++q) a += goals[s * H4_ + q] * P.g2dW[q * H4_ + j];
      Gw[w] = a > 0.f ? a : 0.f;
    }
    __syncthreads();
    for (int tp = 0; tp < DEC_; ++tp) {  // decoder attention contexts, all steps
      if (tid < DEC_) {
        float a = P.daB[0];
        if (tid >= tp)
          for (int j = 0; j < H4_; ++j) a += tanhf(Gw[tid * H4_ + j]) * P.daW[j];
        sc[tid] = a > 0.f ? a : 0.f;
      }
      __syncthreads();
      softmax45(sc, red, tid);
      if (tid < H4_) {
        float a = 0.f;
        for (int s = tp; s < DEC_; ++s) a += sc[s] * Gw[s * H4_ + tid];
        P.gatt[(size_t)((t * DEC_ + tp) * B_ + b) * H4_ + tid] = a;
      }
      __syncthreads();
    }
    {  // dec_hidden0
      float a = P.e2dB[tid];
      for (int q = 0; q < H_; ++q) a += trajh[q] * P.e2dW[q * H_ + tid];
      dh0[tid] = a > 0.f ? a : 0.f;
    }
    __syncthreads();
    if (tid < 2 * L_) {  // prior -> mu, logvar
      float a = P.prB[tid];
      for (int q = 0; q < H_; ++q) a += dh0[q] * P.prW[q * 2 * L_ + tid];
      gnew[tid] = a;
    }
    __syncthreads();
    if (tid < L_) { muv[tid] = gnew[tid]; sigv[tid] = __expf(0.5f * gnew[L_ + tid]); }
    __syncthreads();
    {  // hidden part of cvae2dec
      float a = P.cvB[tid];
      for (int q = 0; q < H_; ++q) a += dh0[q] * P.cvW[q * H_ + tid];
      hpart[tid] = a;
    }
    __syncthreads();
    if (tid < K_ * L_) {
      int k = tid >> 5, l = tid & 31;
      zk[tid] = muv[l] + sigv[l] * P.eps[(size_t)((b * ENC_ + t) * K_ + k) * L_ + l];
    }
    __syncthreads();
    for (int w = tid; w < K_ * H_; w += 256) {
      int k = w >> 8, j = w & 255;
      float a = hpart[j];
      for (int l = 0; l < L_; ++l) a += zk[k * L_ + l] * P.cvW[(size_t)(H_ + l) * H_ + j];
      P.dech0[(size_t)(t * 640 + b * K_ + k) * H_ + j] = a > 0.f ? a : 0.f;
    }
    __syncthreads();
  }
}

// ---------------- K3: WMMA decoder rollout. Grid (20 row-tiles, 15 enc steps) ----------
__global__ void __launch_bounds__(256)
k3_decoder(const _Float16* __restrict__ fr, const float* __restrict__ ga,
           const float* __restrict__ gatt, const float* __restrict__ dech0,
           const float* __restrict__ decWih, const float* __restrict__ dh2iB,
           const float* __restrict__ decBhh, const float* __restrict__ regW,
           const float* __restrict__ regB, float* __restrict__ out) {
  __shared__ _Float16 sDH[2][32 * 256];   // hidden state double buffer (f16)
  __shared__ _Float16 sDDI[32 * 256];     // relu(dh @ dh2i) stage (f16)
  __shared__ float    sGP[3 * 768];       // per-b broadcast: g@Wih[0:64] + ga
  int tile = blockIdx.x, t = blockIdx.y;
  int tid = threadIdx.x, lane = tid & 31, wave = tid >> 5;
  int rowBase = tile * 32;
  int bFirst = rowBase / K_;
  for (int i = tid; i < 32 * 256; i += 256)
    sDH[0][i] = (_Float16)dech0[(size_t)(t * 640 + rowBase + (i >> 8)) * 256 + (i & 255)];
  __syncthreads();
  int cur = 0;
  for (int tau = 0; tau < DEC_; ++tau) {
    // broadcast gate-input part for the (<=3) batch elements this tile touches
    for (int w = tid; w < 3 * 768; w += 256) {
      int s = w / 768, col = w - s * 768;
      int b = bFirst + s;
      float acc = 0.f;
      if (b < B_) {
        const float* g = gatt + (size_t)((t * DEC_ + tau) * B_ + b) * H4_;
        #pragma unroll 8
        for (int j = 0; j < H4_; ++j) acc += g[j] * decWih[(size_t)j * 768 + col];
        acc += ga[(size_t)(t * B_ + b) * 768 + col];
      }
      sGP[w] = acc;
    }
    __syncthreads();
    // ddi = relu(dh @ dh2i + b) : 2 m-tiles x 16 n-tiles over 8 waves
    for (int pi = 0; pi < 4; ++pi) {
      int p = (wave << 2) + pi;
      int mt = p & 1, nt = p >> 1;
      v8f acc = {};
      for (int kt = 0; kt < 8; ++kt) {
        v16h a = load_a_frag(sDH[cur], mt * 16, kt * 32, 256, lane);
        v16h bf = *(const v16h*)(fr + F_DH2I + (size_t)(kt * 16 + nt) * 512 + lane * 16);
        acc = wmma_f16(a, bf, acc);
      }
      int n = (nt << 4) + (lane & 15);
      float bias = dh2iB[n];
      int mOff = (mt << 4) + ((lane >> 4) << 3);
      #pragma unroll
      for (int i = 0; i < 8; ++i) {
        float v = acc[i] + bias;
        sDDI[(mOff + i) * 256 + n] = (_Float16)(v > 0.f ? v : 0.f);
      }
    }
    __syncthreads();
    // gates: gi = ddi@Wih[64:320] + sGP ; gh = dh@Whh + bhh ; GRU update
    for (int pi = 0; pi < 4; ++pi) {
      int p = (wave << 2) + pi;
      int mt = p & 1, ht = p >> 1;
      v8f ir = {}, iz = {}, in_ = {}, hr = {}, hz = {}, hn = {};
      for (int kt = 0; kt < 8; ++kt) {
        v16h aD = load_a_frag(sDDI, mt * 16, kt * 32, 256, lane);
        v16h aH = load_a_frag(sDH[cur], mt * 16, kt * 32, 256, lane);
        const _Float16* wih = fr + F_WIH + (size_t)(kt * 48) * 512 + lane * 16;
        const _Float16* whh = fr + F_WHH + (size_t)(kt * 48) * 512 + lane * 16;
        ir  = wmma_f16(aD, *(const v16h*)(wih + (size_t)(ht) * 512), ir);
        iz  = wmma_f16(aD, *(const v16h*)(wih + (size_t)(16 + ht) * 512), iz);
        in_ = wmma_f16(aD, *(const v16h*)(wih + (size_t)(32 + ht) * 512), in_);
        hr  = wmma_f16(aH, *(const v16h*)(whh + (size_t)(ht) * 512), hr);
        hz  = wmma_f16(aH, *(const v16h*)(whh + (size_t)(16 + ht) * 512), hz);
        hn  = wmma_f16(aH, *(const v16h*)(whh + (size_t)(32 + ht) * 512), hn);
      }
      int n = lane & 15, col = (ht << 4) + n;
      int mOff = (mt << 4) + ((lane >> 4) << 3);
      float br = decBhh[col], bz = decBhh[256 + col], bn = decBhh[512 + col];
      #pragma unroll
      for (int i = 0; i < 8; ++i) {
        int row = mOff + i;
        int s = (rowBase + row) / K_ - bFirst;
        float r = sigmf(ir[i] + sGP[s * 768 + col] + hr[i] + br);
        float z = sigmf(iz[i] + sGP[s * 768 + 256 + col] + hz[i] + bz);
        float nn = tanhf(in_[i] + sGP[s * 768 + 512 + col] + r * (hn[i] + bn));
        float hp = (float)sDH[cur][row * 256 + col];
        sDH[cur ^ 1][row * 256 + col] = (_Float16)((1.f - z) * nn + z * hp);
      }
    }
    __syncthreads();
    cur ^= 1;
    if (tid < 128) {  // traj = tanh(dh @ regressor + b)
      int row = tid >> 2, c = tid & 3;
      float acc = regB[c];
      for (int j = 0; j < 256; ++j) acc += (float)sDH[cur][row * 256 + j] * regW[j * 4 + c];
      int rowg = rowBase + row;
      int b = rowg / K_, k = rowg - b * K_;
      out[OUT_CVAE + (size_t)(((b * ENC_ + t) * DEC_ + tau) * K_ + k) * 4 + c] = tanhf(acc);
    }
    __syncthreads();
  }
}

extern "C" void kernel_launch(void* const* d_in, const int* in_sizes, int n_in,
                              void* d_out, int out_size, void* d_ws, size_t ws_size,
                              hipStream_t stream) {
  (void)in_sizes; (void)n_in; (void)out_size; (void)ws_size;
  // pytree order: inputs, input_angle, z_noise, then params sorted by key
  const float* inputs   = (const float*)d_in[0];
  const float* angle    = (const float*)d_in[1];
  const float* eps      = (const float*)d_in[2];
  const float* addl_Whh = (const float*)d_in[3];  (void)addl_Whh; // h==0 -> unused
  const float* addl_Wih = (const float*)d_in[4];
  const float* addl_bhh = (const float*)d_in[5];
  const float* addl_bih = (const float*)d_in[6];
  const float* cv_W     = (const float*)d_in[7];
  const float* cv_b     = (const float*)d_in[8];
  const float* dec_Whh  = (const float*)d_in[9];
  const float* dec_Wih  = (const float*)d_in[10];
  const float* dec_bhh  = (const float*)d_in[11];
  const float* dec_bih  = (const float*)d_in[12];
  const float* da_W     = (const float*)d_in[13];
  const float* da_b     = (const float*)d_in[14];
  const float* dh2i_W   = (const float*)d_in[15];
  const float* dh2i_b   = (const float*)d_in[16];  (void)dh2i_b;
  const float* e2d_W    = (const float*)d_in[17];
  const float* e2d_b    = (const float*)d_in[18];
  const float* e2g_W    = (const float*)d_in[19];
  const float* e2g_b    = (const float*)d_in[20];
  const float* ea_W     = (const float*)d_in[21];
  const float* ea_b     = (const float*)d_in[22];
  const float* fe_W     = (const float*)d_in[23];
  const float* fe_b     = (const float*)d_in[24];
  const float* g2d_W    = (const float*)d_in[25];
  const float* g2d_b    = (const float*)d_in[26];
  const float* g2e_W    = (const float*)d_in[27];
  const float* g2e_b    = (const float*)d_in[28];
  const float* g2t_W    = (const float*)d_in[29];
  const float* g2t_b    = (const float*)d_in[30];
  const float* gh2i_W   = (const float*)d_in[31];
  const float* gh2i_b   = (const float*)d_in[32];
  const float* goal_Whh = (const float*)d_in[33];
  const float* goal_Wih = (const float*)d_in[34];
  const float* goal_bhh = (const float*)d_in[35];
  const float* goal_bih = (const float*)d_in[36];
  const float* pose_W   = (const float*)d_in[37];
  const float* pose_b   = (const float*)d_in[38];
  const float* pr_W     = (const float*)d_in[39];
  const float* pr_b     = (const float*)d_in[40];
  const float* reg_W    = (const float*)d_in[41];
  const float* reg_b    = (const float*)d_in[42];
  const float* te_Whh   = (const float*)d_in[43];
  const float* te_Wih   = (const float*)d_in[44];
  const float* te_bhh   = (const float*)d_in[45];
  const float* te_bih   = (const float*)d_in[46];

  _Float16* fr = (_Float16*)d_ws;
  float* wsf    = (float*)((char*)d_ws + (size_t)F_TOTAL * sizeof(_Float16));
  float* trajin = wsf + WS_TRAJIN;
  float* addl   = wsf + WS_ADDL;
  float* ga     = wsf + WS_GA;
  float* gatt   = wsf + WS_GATT;
  float* dech0  = wsf + WS_DECH0;
  float* out    = (float*)d_out;

  k0_fragw<<<256, 256, 0, stream>>>(dh2i_W, dec_Wih, dec_Whh, fr);
  k1_features<<<B_ * ENC_, 256, 0, stream>>>(inputs, angle, fe_W, fe_b, pose_W, pose_b,
                                             addl_Wih, addl_bih, addl_bhh, dec_Wih, dec_bih,
                                             trajin, addl, ga, out);
  EncP P;
  P.teWih = te_Wih; P.teWhh = te_Whh; P.teBih = te_bih; P.teBhh = te_bhh;
  P.e2gW = e2g_W; P.e2gB = e2g_b;
  P.gWih = goal_Wih; P.gWhh = goal_Whh; P.gBih = goal_bih; P.gBhh = goal_bhh;
  P.gh2iW = gh2i_W; P.gh2iB = gh2i_b;
  P.g2tW = g2t_W; P.g2tB = g2t_b;
  P.regW = reg_W; P.regB = reg_b;
  P.g2eW = g2e_W; P.g2eB = g2e_b; P.eaW = ea_W; P.eaB = ea_b;
  P.g2dW = g2d_W; P.g2dB = g2d_b; P.daW = da_W; P.daB = da_b;
  P.e2dW = e2d_W; P.e2dB = e2d_b;
  P.prW = pr_W; P.prB = pr_b;
  P.cvW = cv_W; P.cvB = cv_b;
  P.eps = eps; P.trajin = trajin;
  P.gatt = gatt; P.dech0 = dech0; P.out = out;
  k2_encoder<<<B_, 256, 0, stream>>>(P);
  k3_decoder<<<dim3(20, ENC_), 256, 0, stream>>>(fr, ga, gatt, dech0, dec_Wih,
                                                 dh2i_b, dec_bhh, reg_W, reg_b, out);
}
